// ISATconvolution_PM_50027779064028
// MI455X (gfx1250) — matrix operand order repaired and checked by reading, not compiled, not run on hardware
//
#include <hip/hip_runtime.h>
#include <hip/hip_bf16.h>
#include <cstddef>

typedef __attribute__((ext_vector_type(16))) __bf16 v16bf_t;
typedef __attribute__((ext_vector_type(8)))  float  v8f_t;

// All dims are exact multiples of the 16-row WMMA tile -> no tail guards.
#define N_R   200000
#define N_I   200000
#define N_D   40000
#define E_RR  800000
#define E_II  800000
#define E_DD  160000
#define FDIM  128
#define FEDIM 16
#define NLAYER 3

// ---------------------------------------------------------------------------
// WMMA fragment helpers (layouts per CDNA5 ISA 7.12.2, wave32)
// ---------------------------------------------------------------------------

// A: 16x32 bf16 from LDS, row M = lane%16; per lane two contiguous 8-element
// runs (K+0..7 and K+16..23, shifted by 8 for lanes 16-31) -> 2x ds_load_b128.
__device__ __forceinline__ v16bf_t wmma_load_a(const __bf16* As, int ldk, int k0, int lane) {
  const int m = lane & 15, h = lane >> 4;
  const __bf16* row = As + m * ldk + k0 + 8 * h;
  v16bf_t a;
#pragma unroll
  for (int j = 0; j < 8; ++j) {
    a[j]     = row[j];
    a[8 + j] = row[16 + j];
  }
  return a;
}

// B fragments pre-swizzled in global memory: frag (kstep,ntile) stored as
// 32 lanes x 16 bf16 contiguous -> one aligned 32-byte vector load per wave.
__device__ __forceinline__ v16bf_t wmma_load_b_swz(const __bf16* Bsw, int nNtiles,
                                                   int kstep, int ntile, int lane) {
  return *(const v16bf_t*)(Bsw + (((size_t)kstep * nNtiles + ntile) * 32 + lane) * 16);
}

__device__ __forceinline__ v8f_t wmma_bf16(v16bf_t a, v16bf_t b, v8f_t c) {
  return __builtin_amdgcn_wmma_f32_16x16x32_bf16(false, a, false, b, (short)0, c, false, false);
}

__device__ __forceinline__ void store_bf16x4(__bf16* d, float4 v) {
  d[0] = (__bf16)v.x; d[1] = (__bf16)v.y; d[2] = (__bf16)v.z; d[3] = (__bf16)v.w;
}

// ---------------------------------------------------------------------------
// Fused edge-message GEMM + ReLU + scatter-add:
//   agg[dst[e]] += relu(concat(r_node[src[e]], edge[e]) @ We + be)
// K = 144 padded to 160.  16 edges per block, 8 waves (N=128).
// ---------------------------------------------------------------------------
__global__ __launch_bounds__(256) void msg_gemm_scatter(
    const float* __restrict__ rnode, const float* __restrict__ edgef,
    const int* __restrict__ esrc, const int* __restrict__ edst,
    const __bf16* __restrict__ We, const float* __restrict__ be,
    float* __restrict__ agg) {
  __shared__ __bf16 As[16 * 160];
  __shared__ int sSrc[16];
  __shared__ int sDst[16];
  const int ebase = blockIdx.x * 16;
  if (threadIdx.x < 16) {
    sSrc[threadIdx.x] = esrc[ebase + threadIdx.x];
    sDst[threadIdx.x] = edst[ebase + threadIdx.x];
  }
  __syncthreads();
  for (int c = threadIdx.x; c < 16 * 40; c += 256) {  // float4 chunks, 160/4 = 40 per row
    int r = c / 40, kq = (c - r * 40) * 4;
    float4 v = make_float4(0.f, 0.f, 0.f, 0.f);
    if (kq < FDIM)              v = *(const float4*)(rnode + ((size_t)sSrc[r] << 7) + kq);
    else if (kq < FDIM + FEDIM) v = *(const float4*)(edgef + (size_t)(ebase + r) * FEDIM + (kq - FDIM));
    store_bf16x4(As + r * 160 + kq, v);
  }
  __syncthreads();
  const int wave = threadIdx.x >> 5, lane = threadIdx.x & 31;
  v8f_t acc = {};
#pragma unroll
  for (int ks = 0; ks < 5; ++ks)
    acc = wmma_bf16(wmma_load_a(As, 160, ks * 32, lane),
                    wmma_load_b_swz(We, 8, ks, wave, lane), acc);
  const int n = lane & 15, h = lane >> 4, n0 = wave * 16;
  const float bia = be[n0 + n];
#pragma unroll
  for (int v = 0; v < 8; ++v) {
    float x = acc[v] + bia;
    x = x > 0.f ? x : 0.f;
    atomicAdd(&agg[((size_t)sDst[v + 8 * h] << 7) + n0 + n], x);
  }
}

// ---------------------------------------------------------------------------
// Fused node update: r = relu(concat(r, agg) @ Wn + bn); i_P += r; i_M += r
// K = 256.  Row-wise -> in-place safe (rows staged to LDS before writes).
// ---------------------------------------------------------------------------
__global__ __launch_bounds__(256) void node_update_gemm(
    float* __restrict__ rnode, const float* __restrict__ agg,
    const __bf16* __restrict__ Wn, const float* __restrict__ bn,
    float* __restrict__ iP, float* __restrict__ iM) {
  __shared__ __bf16 As[16 * 256];
  const int rbase = blockIdx.x * 16;
  for (int c = threadIdx.x; c < 16 * 64; c += 256) {  // 256/4 = 64 chunks per row
    int r = c >> 6, kq = (c & 63) * 4;
    int row = rbase + r;
    float4 v = (kq < FDIM) ? *(const float4*)(rnode + ((size_t)row << 7) + kq)
                           : *(const float4*)(agg + ((size_t)row << 7) + (kq - FDIM));
    store_bf16x4(As + r * 256 + kq, v);
  }
  __syncthreads();
  const int wave = threadIdx.x >> 5, lane = threadIdx.x & 31;
  v8f_t acc = {};
#pragma unroll
  for (int ks = 0; ks < 8; ++ks)
    acc = wmma_bf16(wmma_load_a(As, 256, ks * 32, lane),
                    wmma_load_b_swz(Wn, 8, ks, wave, lane), acc);
  const int n = lane & 15, h = lane >> 4, n0 = wave * 16;
  const float bia = bn[n0 + n];
#pragma unroll
  for (int v = 0; v < 8; ++v) {
    float x = acc[v] + bia;
    x = x > 0.f ? x : 0.f;
    size_t o = ((size_t)(rbase + v + 8 * h) << 7) + n0 + n;
    rnode[o] = x;
    iP[o] += x;
    iM[o] += x;
  }
}

// ---------------------------------------------------------------------------
// Generic row-wise GEMM: out = act((A0 [+ A1]) @ B + bias), K = 128.
// Nout in {64,128}; blockDim = 32 * (Nout/16).  act: 0 relu, 1 leaky(0.01).
// ---------------------------------------------------------------------------
__global__ __launch_bounds__(256) void rowsum_gemm(
    const float* __restrict__ A0, const float* __restrict__ A1,
    const __bf16* __restrict__ B, const float* __restrict__ bias,
    float* __restrict__ out, int Nout, int act) {
  __shared__ __bf16 As[16 * 128];
  const int rbase = blockIdx.x * 16;
  const int nNtiles = Nout >> 4;
  for (int c = threadIdx.x; c < 16 * 32; c += blockDim.x) {  // 128/4 = 32 chunks per row
    int r = c >> 5, kq = (c & 31) * 4;
    int row = rbase + r;
    float4 v = *(const float4*)(A0 + ((size_t)row << 7) + kq);
    if (A1) {
      float4 w = *(const float4*)(A1 + ((size_t)row << 7) + kq);
      v.x += w.x; v.y += w.y; v.z += w.z; v.w += w.w;
    }
    store_bf16x4(As + r * 128 + kq, v);
  }
  __syncthreads();
  const int wave = threadIdx.x >> 5, lane = threadIdx.x & 31;
  v8f_t acc = {};
#pragma unroll
  for (int ks = 0; ks < 4; ++ks)
    acc = wmma_bf16(wmma_load_a(As, 128, ks * 32, lane),
                    wmma_load_b_swz(B, nNtiles, ks, wave, lane), acc);
  const int n = lane & 15, h = lane >> 4, n0 = wave * 16;
  const float bia = bias[n0 + n];
#pragma unroll
  for (int v = 0; v < 8; ++v) {
    float x = acc[v] + bia;
    x = (x > 0.f) ? x : ((act == 1) ? 0.01f * x : 0.f);
    out[(size_t)(rbase + v + 8 * h) * Nout + n0 + n] = x;
  }
}

// ---------------------------------------------------------------------------
// Elementwise / scatter helpers
// ---------------------------------------------------------------------------
__global__ void zero_f32(float* __restrict__ p, long long n) {
  long long i = (long long)blockIdx.x * blockDim.x + threadIdx.x;
  if (i < n) p[i] = 0.f;
}

// Convert fp32 weight [Ksrc x Ncol] (zero-pad K to Kdst) directly into the
// swizzled WMMA-B fragment layout: element idx = ((kstep*nNtiles+ntile)*32 + lane)*16 + j
// holds B[kstep*32 + 8*(lane>>4) + (j&7) + 16*(j>>3)][ntile*16 + (lane&15)].
__global__ void convert_swizzle(const float* __restrict__ src, __bf16* __restrict__ dst,
                                int Ksrc, int Kdst, int Ncol) {
  int idx = blockIdx.x * 256 + threadIdx.x;
  if (idx >= Kdst * Ncol) return;
  int j    = idx & 15;
  int lane = (idx >> 4) & 31;
  int f    = idx >> 9;
  int nNtiles = Ncol >> 4;
  int kstep = f / nNtiles, ntile = f - kstep * nNtiles;
  int k = kstep * 32 + 8 * (lane >> 4) + (j & 7) + 16 * (j >> 3);
  int n = ntile * 16 + (lane & 15);
  dst[idx] = (__bf16)((k < Ksrc) ? src[(size_t)k * Ncol + n] : 0.f);
}

// out[dst[e]][f] += x[(src?src[e]:e)][f]   (F = 128)
__global__ void scatter_add128(const float* __restrict__ x, const int* __restrict__ src,
                               const int* __restrict__ dst, float* __restrict__ out,
                               long long total) {
  long long idx = (long long)blockIdx.x * blockDim.x + threadIdx.x;
  if (idx >= total) return;
  long long e = idx >> 7; int f = (int)(idx & 127);
  int s = src ? src[e] : (int)e;
  atomicAdd(&out[((size_t)dst[e] << 7) + f], x[((size_t)s << 7) + f]);
}

// out[dst[e]][f] += 0.1^dist[e] * x[src[e]][f]
__global__ void scatter_dec128(const float* __restrict__ x, const int* __restrict__ src,
                               const int* __restrict__ dst, const int* __restrict__ dist,
                               float* __restrict__ out, long long total) {
  long long idx = (long long)blockIdx.x * blockDim.x + threadIdx.x;
  if (idx >= total) return;
  long long e = idx >> 7; int f = (int)(idx & 127);
  int dd = dist[e] & 3;
  float dec = (dd == 0) ? 1.0f : ((dd == 1) ? 0.1f : ((dd == 2) ? 0.01f : 0.001f));
  atomicAdd(&out[((size_t)dst[e] << 7) + f], dec * x[((size_t)src[e] << 7) + f]);
}

// tiny MLP tail: s2 = leaky(s1 @ W2[64x3] + b2); s = s2 @ W3[3x1] + b3
// plus block-reduced sum / sumsq for the BatchNorm.
__global__ __launch_bounds__(256) void mlp_tail(
    const float* __restrict__ s1, const float* __restrict__ W2, const float* __restrict__ b2,
    const float* __restrict__ W3, const float* __restrict__ b3,
    float* __restrict__ sraw, float* __restrict__ accum, int M) {
  __shared__ float red[256], red2[256];
  int i = blockIdx.x * 256 + threadIdx.x;
  float s = 0.f;
  if (i < M) {
    float s2[3];
#pragma unroll
    for (int j = 0; j < 3; ++j) {
      float a = b2[j];
      for (int k = 0; k < 64; ++k) a += s1[((size_t)i << 6) + k] * W2[k * 3 + j];
      s2[j] = a > 0.f ? a : 0.01f * a;
    }
    s = b3[0];
#pragma unroll
    for (int j = 0; j < 3; ++j) s += s2[j] * W3[j];
    sraw[i] = s;
  }
  red[threadIdx.x]  = (i < M) ? s : 0.f;
  red2[threadIdx.x] = (i < M) ? s * s : 0.f;
  __syncthreads();
  for (int st = 128; st > 0; st >>= 1) {
    if (threadIdx.x < st) {
      red[threadIdx.x]  += red[threadIdx.x + st];
      red2[threadIdx.x] += red2[threadIdx.x + st];
    }
    __syncthreads();
  }
  if (threadIdx.x == 0) {
    atomicAdd(&accum[0], red[0]);
    atomicAdd(&accum[1], red2[0]);
  }
}

__global__ void bn_sigmoid(const float* __restrict__ sraw, const float* __restrict__ accum,
                           const float* __restrict__ g, const float* __restrict__ beta,
                           float* __restrict__ sout, int M) {
  int i = blockIdx.x * 256 + threadIdx.x;
  if (i >= M) return;
  float mu  = accum[0] / (float)M;
  float var = accum[1] / (float)M - mu * mu;
  float x = (sraw[i] - mu) * rsqrtf(var + 1e-5f) * g[0] + beta[0];
  sout[i] = 1.f / (1.f + expf(-x));
}

// out[r][f] = s[d2r_src[r]] * rnode[r][f]
__global__ void gate_scale(const float* __restrict__ s, const int* __restrict__ d2r,
                           const float* __restrict__ rn, float* __restrict__ o, long long total) {
  long long idx = (long long)blockIdx.x * blockDim.x + threadIdx.x;
  if (idx >= total) return;
  o[idx] = s[d2r[idx >> 7]] * rn[idx];
}

// ---------------------------------------------------------------------------
// Host orchestration
// ---------------------------------------------------------------------------
extern "C" void kernel_launch(void* const* d_in, const int* in_sizes, int n_in,
                              void* d_out, int out_size, void* d_ws, size_t ws_size,
                              hipStream_t stream) {
  const float* r_node   = (const float*)d_in[0];
  const float* i_node   = (const float*)d_in[1];
  const float* r2r_edge = (const float*)d_in[2];
  const int* r2r_src = (const int*)d_in[3];
  const int* r2r_dst = (const int*)d_in[4];
  const int* i2i_src = (const int*)d_in[5];
  const int* i2i_dst = (const int*)d_in[6];
  const int* dd_src  = (const int*)d_in[7];
  const int* dd_dst  = (const int*)d_in[8];
  const int* dd_dist = (const int*)d_in[9];
  const int* i2d_dst = (const int*)d_in[11];   // i2d_src == arange
  const int* d2r_src = (const int*)d_in[12];   // d2r_dst == arange
  const float* We = (const float*)d_in[15];
  const float* be = (const float*)d_in[16];
  const float* Wn = (const float*)d_in[17];
  const float* bn = (const float*)d_in[18];
  const float* WiP = (const float*)d_in[19];
  const float* biP = (const float*)d_in[20];
  const float* WiM = (const float*)d_in[21];
  const float* biM = (const float*)d_in[22];
  const float* WdP = (const float*)d_in[23];
  const float* bdP = (const float*)d_in[24];
  const float* WdM = (const float*)d_in[25];
  const float* bdM = (const float*)d_in[26];
  const float* scP_W1 = (const float*)d_in[27];
  const float* scP_b1 = (const float*)d_in[28];
  const float* scP_W2 = (const float*)d_in[29];
  const float* scP_b2 = (const float*)d_in[30];
  const float* scP_W3 = (const float*)d_in[31];
  const float* scP_b3 = (const float*)d_in[32];
  const float* scM_W1 = (const float*)d_in[33];
  const float* scM_b1 = (const float*)d_in[34];
  const float* scM_W2 = (const float*)d_in[35];
  const float* scM_b2 = (const float*)d_in[36];
  const float* scM_W3 = (const float*)d_in[37];
  const float* scM_b3 = (const float*)d_in[38];
  const float* bnP_g = (const float*)d_in[39];
  const float* bnP_b = (const float*)d_in[40];
  const float* bnM_g = (const float*)d_in[41];
  const float* bnM_b = (const float*)d_in[42];
  float* out = (float*)d_out;

  char* base = (char*)d_ws;
  size_t off = 0;
  auto carve = [&](size_t bytes) -> char* {
    char* p = base + off;
    off += (bytes + 255) & ~(size_t)255;
    return p;
  };
  __bf16* We_bf  = (__bf16*)carve((size_t)NLAYER * 160 * 128 * 2);
  __bf16* Wn_bf  = (__bf16*)carve((size_t)NLAYER * 256 * 128 * 2);
  __bf16* WiP_bf = (__bf16*)carve((size_t)NLAYER * 128 * 128 * 2);
  __bf16* WiM_bf = (__bf16*)carve((size_t)NLAYER * 128 * 128 * 2);
  __bf16* WdP_bf = (__bf16*)carve((size_t)128 * 128 * 2);
  __bf16* WdM_bf = (__bf16*)carve((size_t)128 * 128 * 2);
  __bf16* W1P_bf = (__bf16*)carve((size_t)128 * 64 * 2);
  __bf16* W1M_bf = (__bf16*)carve((size_t)128 * 64 * 2);
  float* r_cur = (float*)carve((size_t)N_R * FDIM * 4);
  float* agg   = (float*)carve((size_t)N_R * FDIM * 4);
  float* iP    = (float*)carve((size_t)N_I * FDIM * 4);
  float* iM    = (float*)carve((size_t)N_I * FDIM * 4);
  float* tmp   = (float*)carve((size_t)N_I * FDIM * 4);
  float* dP    = (float*)carve((size_t)N_D * FDIM * 4);
  float* dM    = (float*)carve((size_t)N_D * FDIM * 4);
  float* dagg  = (float*)carve((size_t)N_D * FDIM * 4);
  float* scoreb= (float*)carve((size_t)N_D * FDIM * 4);
  float* s1    = (float*)carve((size_t)N_D * 64 * 4);
  float* srawP = (float*)carve((size_t)N_D * 4);
  float* srawM = (float*)carve((size_t)N_D * 4);
  float* accum = (float*)carve(4 * sizeof(float));

  // --- weight conversion to swizzled bf16 fragments (padded K where needed) ---
  for (int l = 0; l < NLAYER; ++l) {
    convert_swizzle<<<(160 * 128 + 255) / 256, 256, 0, stream>>>(
        We + (size_t)l * 144 * 128, We_bf + (size_t)l * 160 * 128, 144, 160, 128);
    convert_swizzle<<<(256 * 128 + 255) / 256, 256, 0, stream>>>(
        Wn + (size_t)l * 256 * 128, Wn_bf + (size_t)l * 256 * 128, 256, 256, 128);
    convert_swizzle<<<(128 * 128 + 255) / 256, 256, 0, stream>>>(
        WiP + (size_t)l * 128 * 128, WiP_bf + (size_t)l * 128 * 128, 128, 128, 128);
    convert_swizzle<<<(128 * 128 + 255) / 256, 256, 0, stream>>>(
        WiM + (size_t)l * 128 * 128, WiM_bf + (size_t)l * 128 * 128, 128, 128, 128);
  }
  convert_swizzle<<<(128 * 128 + 255) / 256, 256, 0, stream>>>(WdP, WdP_bf, 128, 128, 128);
  convert_swizzle<<<(128 * 128 + 255) / 256, 256, 0, stream>>>(WdM, WdM_bf, 128, 128, 128);
  convert_swizzle<<<(128 * 64 + 255) / 256, 256, 0, stream>>>(scP_W1, W1P_bf, 128, 128, 64);
  convert_swizzle<<<(128 * 64 + 255) / 256, 256, 0, stream>>>(scM_W1, W1M_bf, 128, 128, 64);

  // --- working copies (inputs must not be mutated) ---
  hipMemcpyAsync(r_cur, r_node, (size_t)N_R * FDIM * 4, hipMemcpyDeviceToDevice, stream);
  hipMemcpyAsync(iP, i_node, (size_t)N_I * FDIM * 4, hipMemcpyDeviceToDevice, stream);
  hipMemcpyAsync(iM, i_node, (size_t)N_I * FDIM * 4, hipMemcpyDeviceToDevice, stream);

  const long long nrf = (long long)N_R * FDIM;
  const long long nif = (long long)N_I * FDIM;
  const long long ndf = (long long)N_D * FDIM;

  // --- 3 message-passing layers ---
  for (int l = 0; l < NLAYER; ++l) {
    zero_f32<<<(unsigned)((nrf + 255) / 256), 256, 0, stream>>>(agg, nrf);
    msg_gemm_scatter<<<E_RR / 16, 256, 0, stream>>>(
        r_cur, r2r_edge, r2r_src, r2r_dst,
        We_bf + (size_t)l * 160 * 128, be + (size_t)l * 128, agg);
    node_update_gemm<<<N_R / 16, 256, 0, stream>>>(
        r_cur, agg, Wn_bf + (size_t)l * 256 * 128, bn + (size_t)l * 128, iP, iM);

    zero_f32<<<(unsigned)((nif + 255) / 256), 256, 0, stream>>>(tmp, nif);
    scatter_add128<<<(unsigned)(((long long)E_II * 128 + 255) / 256), 256, 0, stream>>>(
        iP, i2i_src, i2i_dst, tmp, (long long)E_II * 128);
    rowsum_gemm<<<N_I / 16, 256, 0, stream>>>(
        iP, tmp, WiP_bf + (size_t)l * 128 * 128, biP + (size_t)l * 128, iP, 128, 0);

    zero_f32<<<(unsigned)((nif + 255) / 256), 256, 0, stream>>>(tmp, nif);
    scatter_add128<<<(unsigned)(((long long)E_II * 128 + 255) / 256), 256, 0, stream>>>(
        iM, i2i_src, i2i_dst, tmp, (long long)E_II * 128);
    rowsum_gemm<<<N_I / 16, 256, 0, stream>>>(
        iM, tmp, WiM_bf + (size_t)l * 128 * 128, biM + (size_t)l * 128, iM, 128, 0);
  }

  // --- i2d aggregation ---
  zero_f32<<<(unsigned)((ndf + 255) / 256), 256, 0, stream>>>(dP, ndf);
  scatter_add128<<<(unsigned)((nif + 255) / 256), 256, 0, stream>>>(iP, nullptr, i2d_dst, dP, nif);
  zero_f32<<<(unsigned)((ndf + 255) / 256), 256, 0, stream>>>(dM, ndf);
  scatter_add128<<<(unsigned)((nif + 255) / 256), 256, 0, stream>>>(iM, nullptr, i2d_dst, dM, nif);

  float* sP_out = out + (size_t)2 * N_R * FDIM;
  float* sM_out = sP_out + N_D;

  // --- score branch P ---
  zero_f32<<<(unsigned)((ndf + 255) / 256), 256, 0, stream>>>(dagg, ndf);
  scatter_dec128<<<(unsigned)(((long long)E_DD * 128 + 255) / 256), 256, 0, stream>>>(
      dP, dd_src, dd_dst, dd_dist, dagg, (long long)E_DD * 128);
  rowsum_gemm<<<N_D / 16, 256, 0, stream>>>(dP, dagg, WdP_bf, bdP, scoreb, 128, 0);
  rowsum_gemm<<<N_D / 16, 128, 0, stream>>>(scoreb, nullptr, W1P_bf, scP_b1, s1, 64, 1);
  zero_f32<<<1, 256, 0, stream>>>(accum, 4);
  mlp_tail<<<(N_D + 255) / 256, 256, 0, stream>>>(s1, scP_W2, scP_b2, scP_W3, scP_b3, srawP, accum, N_D);
  bn_sigmoid<<<(N_D + 255) / 256, 256, 0, stream>>>(srawP, accum, bnP_g, bnP_b, sP_out, N_D);

  // --- score branch M ---
  zero_f32<<<(unsigned)((ndf + 255) / 256), 256, 0, stream>>>(dagg, ndf);
  scatter_dec128<<<(unsigned)(((long long)E_DD * 128 + 255) / 256), 256, 0, stream>>>(
      dM, dd_src, dd_dst, dd_dist, dagg, (long long)E_DD * 128);
  rowsum_gemm<<<N_D / 16, 256, 0, stream>>>(dM, dagg, WdM_bf, bdM, scoreb, 128, 0);
  rowsum_gemm<<<N_D / 16, 128, 0, stream>>>(scoreb, nullptr, W1M_bf, scM_b1, s1, 64, 1);
  mlp_tail<<<(N_D + 255) / 256, 256, 0, stream>>>(s1, scM_W2, scM_b2, scM_W3, scM_b3, srawM, accum + 2, N_D);
  bn_sigmoid<<<(N_D + 255) / 256, 256, 0, stream>>>(srawM, accum + 2, bnM_g, bnM_b, sM_out, N_D);

  // --- d2r gating (d2r_dst == arange -> pure gather) ---
  gate_scale<<<(unsigned)((nrf + 255) / 256), 256, 0, stream>>>(sP_out, d2r_src, r_cur, out, nrf);
  gate_scale<<<(unsigned)((nrf + 255) / 256), 256, 0, stream>>>(sM_out, d2r_src, r_cur, out + nrf, nrf);
}
